// MultiTaskModel_34445637714048
// MI455X (gfx1250) — matrix operand order repaired and checked by reading, not compiled, not run on hardware
//
#include <hip/hip_runtime.h>
#include <hip/hip_bf16.h>
#include <stdint.h>

// ---------------------------------------------------------------------------
// MI455X multi-task MLP, bf16 WMMA path.
//
// Roofline: x read = 157 MB -> 6.7us floor @23.3TB/s. Selected-head-only
// FLOPs = 9.4 GFLOP -> bf16 WMMA keeps compute under the memory floor,
// fp32 WMMA would not. So: stream fp32 x, convert to bf16 in regs,
// v_wmma_f32_16x16x32_bf16 everywhere, fp32 accumulate, bias/relu in fp32.
// Backbone processes 2 M-tiles (32 rows) per wave so each packed-B tile is
// loaded once per 2 WMMAs -> halves L2 B traffic, keeps ~120 VGPRs so
// multiple waves/SIMD still hide HBM latency.
// ---------------------------------------------------------------------------

typedef __bf16  bf16_t;
typedef __bf16  v16bf __attribute__((ext_vector_type(16)));
typedef float   v8f   __attribute__((ext_vector_type(8)));
typedef float   f32x4 __attribute__((ext_vector_type(4)));
typedef unsigned int u32x4 __attribute__((ext_vector_type(4)));

#define NUM_TASKS 8
#define IN_DIM    300
#define HID       64
#define H1D       128
#define H2D       64

__device__ __forceinline__ unsigned short f2bf(float f) {
  union { bf16_t h; unsigned short s; } u;
  u.h = (bf16_t)f;
  return u.s;
}

__device__ __forceinline__ v16bf mk16(u32x4 a, u32x4 b) {
  union { u32x4 u[2]; v16bf v; } w;
  w.u[0] = a; w.u[1] = b;
  return w.v;
}

__device__ __forceinline__ v8f wmma_bf16(v16bf a, v16bf b, v8f c) {
  // v_wmma_f32_16x16x32_bf16  D = A(16x32) * B(32x16) + C
  return __builtin_amdgcn_wmma_f32_16x16x32_bf16(
      false, a, false, b, (short)0, c, false, false);
}

// Build a 16x32 bf16 A tile from one fp32 row (full-tile fast path).
// Per-lane mapping: k = k0 + (e%8) + (e/8)*16, where k0 = kt*32 + (l/16)*8.
__device__ __forceinline__ v16bf load_a_f32(const float* __restrict__ xrow,
                                            int k0) {
  f32x4 A0 = *(const f32x4*)(xrow + k0);
  f32x4 A1 = *(const f32x4*)(xrow + k0 + 4);
  f32x4 B0 = *(const f32x4*)(xrow + k0 + 16);
  f32x4 B1 = *(const f32x4*)(xrow + k0 + 20);
  v16bf a;
  a[0]=(bf16_t)A0.x; a[1]=(bf16_t)A0.y; a[2]=(bf16_t)A0.z; a[3]=(bf16_t)A0.w;
  a[4]=(bf16_t)A1.x; a[5]=(bf16_t)A1.y; a[6]=(bf16_t)A1.z; a[7]=(bf16_t)A1.w;
  a[8]=(bf16_t)B0.x; a[9]=(bf16_t)B0.y; a[10]=(bf16_t)B0.z; a[11]=(bf16_t)B0.w;
  a[12]=(bf16_t)B1.x; a[13]=(bf16_t)B1.y; a[14]=(bf16_t)B1.z; a[15]=(bf16_t)B1.w;
  return a;
}

// Tail A tile with k<IN_DIM guard (zero-padded to K=320).
__device__ __forceinline__ v16bf load_a_f32_tail(const float* __restrict__ xrow,
                                                 int hi) {
  v16bf a;
#pragma unroll
  for (int e = 0; e < 16; ++e) {
    int k = 288 + hi * 8 + (e & 7) + ((e >> 3) << 4);
    float v = (k < IN_DIM) ? xrow[k] : 0.0f;
    a[e] = (bf16_t)v;
  }
  return a;
}

// ---------------------------------------------------------------------------
// Stage 0a: zero task counters
// ---------------------------------------------------------------------------
__global__ void k_zero_cnt(unsigned int* cnt) {
  if (threadIdx.x < NUM_TASKS) cnt[threadIdx.x] = 0u;
}

// ---------------------------------------------------------------------------
// Stage 0b: pack fp32 weights into bf16 WMMA B-matrix tiles.
// Tile (kt,nt) is 512 bf16: lane-major, 16 elems/lane.
//   n = nt*16 + lane%16 ;  k = kt*32 + (lane/16)*8 + (e%8) + (e/8)*16    (e = elem 0..15)
// Out-of-range (k>=K or n>=Ncols) -> 0  (pads K 300->320, W3 N 1->16).
// ---------------------------------------------------------------------------
__global__ void k_pack_b(const float* __restrict__ src,
                         unsigned short* __restrict__ dst,
                         int K, int Ncols, int Ktiles, int Ntiles, int batch,
                         long srcStride, long dstStride) {
  long idx = (long)blockIdx.x * blockDim.x + threadIdx.x;
  long per = (long)Ktiles * Ntiles * 512;
  long tot = per * batch;
  if (idx >= tot) return;
  int  b    = (int)(idx / per);
  long rem  = idx % per;
  int  tile = (int)(rem >> 9);
  int  w    = (int)(rem & 511);
  int  lane = w >> 4, e = w & 15;
  int  kt = tile / Ntiles, nt = tile % Ntiles;
  int  k = kt * 32 + (lane >> 4) * 8 + (e & 7) + ((e >> 3) << 4);
  int  n = nt * 16 + (lane & 15);
  float v = (k < K && n < Ncols) ? src[(long)b * srcStride + (long)k * Ncols + n]
                                 : 0.0f;
  dst[(long)b * dstStride + rem] = f2bf(v);
}

// ---------------------------------------------------------------------------
// Stage 1: bucket rows by task id
// ---------------------------------------------------------------------------
__global__ void k_bin(const int* __restrict__ ids, unsigned int* __restrict__ cnt,
                      unsigned int* __restrict__ bucket, float* __restrict__ out,
                      int n) {
  int i = blockIdx.x * blockDim.x + threadIdx.x;
  if (i >= n) return;
  int t = ids[i];
  if (t >= 0 && t < NUM_TASKS) {
    unsigned s = atomicAdd(&cnt[t], 1u);
    bucket[(long)t * n + s] = (unsigned)i;
  } else {
    out[i] = 0.0f;  // invalid task id -> 0 (reference semantics)
  }
}

// ---------------------------------------------------------------------------
// Stage 2: backbone  feats = relu(x @ Wb + bb), bf16 row-major [N,64]
// One wave per 32 rows (2 M-tiles). Each B tile loaded once, used by 2 WMMAs.
// ---------------------------------------------------------------------------
__global__ __launch_bounds__(32)
void k_backbone(const float* __restrict__ x, const float* __restrict__ bb,
                const unsigned short* __restrict__ wbPack,
                unsigned short* __restrict__ feats, int n) {
  const int l  = threadIdx.x;
  const int m  = l & 15;       // N col within tile / row within tile
  const int hi = l >> 4;       // lane group (K split)
  const int rbase = blockIdx.x * 32;

  int row0 = rbase + m;
  int row1 = rbase + 16 + m;
  int r0c = row0 < n ? row0 : n - 1;
  int r1c = row1 < n ? row1 : n - 1;
  const float* xrow0 = x + (long)r0c * IN_DIM;
  const float* xrow1 = x + (long)r1c * IN_DIM;
  const u32x4* wbp = (const u32x4*)wbPack;

  v8f acc[2][4];
#pragma unroll
  for (int mt = 0; mt < 2; ++mt)
#pragma unroll
    for (int nt = 0; nt < 4; ++nt) acc[mt][nt] = (v8f){};

  // 9 full K-tiles (K 0..287)
  for (int kt = 0; kt < 9; ++kt) {
    int k0 = kt * 32 + hi * 8;
    v16bf a0 = load_a_f32(xrow0, k0);
    v16bf a1 = load_a_f32(xrow1, k0);
#pragma unroll
    for (int nt = 0; nt < 4; ++nt) {
      int bi = (kt * 4 + nt) * 64 + l * 2;
      v16bf bv = mk16(wbp[bi], wbp[bi + 1]);
      acc[0][nt] = wmma_bf16(a0, bv, acc[0][nt]);
      acc[1][nt] = wmma_bf16(a1, bv, acc[1][nt]);
    }
  }
  // tail K-tile (K 288..299, zero padded to 320)
  {
    v16bf a0 = load_a_f32_tail(xrow0, hi);
    v16bf a1 = load_a_f32_tail(xrow1, hi);
#pragma unroll
    for (int nt = 0; nt < 4; ++nt) {
      int bi = (9 * 4 + nt) * 64 + l * 2;
      v16bf bv = mk16(wbp[bi], wbp[bi + 1]);
      acc[0][nt] = wmma_bf16(a0, bv, acc[0][nt]);
      acc[1][nt] = wmma_bf16(a1, bv, acc[1][nt]);
    }
  }

  // bias + relu + store feats bf16 row-major; C layout: M=r+8*hi, N=nt*16+m
#pragma unroll
  for (int mt = 0; mt < 2; ++mt) {
#pragma unroll
    for (int nt = 0; nt < 4; ++nt) {
      int col = nt * 16 + m;
      float bias = bb[col];
#pragma unroll
      for (int r = 0; r < 8; ++r) {
        int rr = rbase + mt * 16 + r + 8 * hi;
        if (rr < n) {
          float v = fmaxf(acc[mt][nt][r] + bias, 0.0f);
          feats[(long)rr * HID + col] = f2bf(v);
        }
      }
    }
  }
}

// ---------------------------------------------------------------------------
// Stage 3: per-task heads on bucketed 16-row groups. One wave per group.
// A for layer1 gathered from feats directly in A layout; LDS round trips
// re-shape C layout -> A layout between layers.
// ---------------------------------------------------------------------------
__global__ __launch_bounds__(32)
void k_heads(const unsigned int* __restrict__ cnt,
             const unsigned int* __restrict__ bucket,
             const unsigned short* __restrict__ feats,
             const unsigned short* __restrict__ w1Pack,
             const unsigned short* __restrict__ w2Pack,
             const unsigned short* __restrict__ w3Pack,
             const float* __restrict__ b1, const float* __restrict__ b2,
             const float* __restrict__ b3, float* __restrict__ out,
             int n, int groups) {
  const int t = blockIdx.x / groups;
  const int g = blockIdx.x % groups;
  const int base = g * 16;
  const int c = (int)cnt[t];
  if (base >= c) return;  // wave-uniform

  __shared__ __attribute__((aligned(16))) int rowsS[16];
  __shared__ __attribute__((aligned(16))) unsigned short h1S[16 * H1D];
  __shared__ __attribute__((aligned(16))) unsigned short h2S[16 * H2D];

  const int l = threadIdx.x;
  const int m = l & 15;
  const int hi = l >> 4;

  if (l < 16) {
    int src = (base + l < c) ? (base + l) : base;  // dup first row as pad
    rowsS[l] = (int)bucket[(long)t * n + src];
  }
  __syncthreads();

  // ---- layer 1: [16,64] @ W1[t] [64,128] ----
  const u32x4* fp = (const u32x4*)(feats + (long)rowsS[m] * HID);
  const u32x4* w1p = (const u32x4*)(w1Pack + (long)t * (2 * 8 * 512));
  v8f c1a[8];
#pragma unroll
  for (int nt = 0; nt < 8; ++nt) c1a[nt] = (v8f){};
#pragma unroll
  for (int kt = 0; kt < 2; ++kt) {
    int k0 = kt * 32 + hi * 8;            // in bf16 elems; /8 -> u32x4 idx
    v16bf a = mk16(fp[k0 >> 3], fp[(k0 >> 3) + 2]);
#pragma unroll
    for (int nt = 0; nt < 8; ++nt) {
      int bi = (kt * 8 + nt) * 64 + l * 2;
      c1a[nt] = wmma_bf16(a, mk16(w1p[bi], w1p[bi + 1]), c1a[nt]);
    }
  }
#pragma unroll
  for (int nt = 0; nt < 8; ++nt) {
    int col = nt * 16 + m;
    float bias = b1[t * H1D + col];
#pragma unroll
    for (int r = 0; r < 8; ++r) {
      float v = fmaxf(c1a[nt][r] + bias, 0.0f);
      h1S[(r + 8 * hi) * H1D + col] = f2bf(v);
    }
  }
  __syncthreads();

  // ---- layer 2: [16,128] @ W2[t] [128,64] ----
  const u32x4* w2p = (const u32x4*)(w2Pack + (long)t * (4 * 4 * 512));
  v8f c2a[4];
#pragma unroll
  for (int nt = 0; nt < 4; ++nt) c2a[nt] = (v8f){};
#pragma unroll
  for (int kt = 0; kt < 4; ++kt) {
    int k0 = kt * 32 + hi * 8;
    v16bf a = mk16(*(const u32x4*)&h1S[m * H1D + k0],
                   *(const u32x4*)&h1S[m * H1D + k0 + 16]);
#pragma unroll
    for (int nt = 0; nt < 4; ++nt) {
      int bi = (kt * 4 + nt) * 64 + l * 2;
      c2a[nt] = wmma_bf16(a, mk16(w2p[bi], w2p[bi + 1]), c2a[nt]);
    }
  }
#pragma unroll
  for (int nt = 0; nt < 4; ++nt) {
    int col = nt * 16 + m;
    float bias = b2[t * H2D + col];
#pragma unroll
    for (int r = 0; r < 8; ++r) {
      float v = fmaxf(c2a[nt][r] + bias, 0.0f);
      h2S[(r + 8 * hi) * H2D + col] = f2bf(v);
    }
  }
  __syncthreads();

  // ---- layer 3: [16,64] @ W3[t] [64,1] (B zero-padded to 16 cols) ----
  const u32x4* w3p = (const u32x4*)(w3Pack + (long)t * (2 * 1 * 512));
  v8f oc = {};
#pragma unroll
  for (int kt = 0; kt < 2; ++kt) {
    int k0 = kt * 32 + hi * 8;
    v16bf a = mk16(*(const u32x4*)&h2S[m * H2D + k0],
                   *(const u32x4*)&h2S[m * H2D + k0 + 16]);
    int bi = kt * 64 + l * 2;
    oc = wmma_bf16(a, mk16(w3p[bi], w3p[bi + 1]), oc);
  }

  // result col 0 lives in lanes with m==0 (lane 0: rows 0..7, lane 16: 8..15)
  if (m == 0) {
    float bias = b3[t];
#pragma unroll
    for (int r = 0; r < 8; ++r) {
      int rr = r + 8 * hi;
      if (base + rr < c) out[rowsS[rr]] = oc[r] + bias;
    }
  }
}

// ---------------------------------------------------------------------------
// Host launcher
// ---------------------------------------------------------------------------
extern "C" void kernel_launch(void* const* d_in, const int* in_sizes, int n_in,
                              void* d_out, int out_size, void* d_ws, size_t ws_size,
                              hipStream_t stream) {
  (void)n_in; (void)out_size; (void)ws_size;
  const float* x   = (const float*)d_in[0];
  const int*   ids = (const int*)d_in[1];
  const float* Wb  = (const float*)d_in[2];
  const float* bb  = (const float*)d_in[3];
  const float* W1  = (const float*)d_in[4];
  const float* b1  = (const float*)d_in[5];
  const float* W2  = (const float*)d_in[6];
  const float* b2  = (const float*)d_in[7];
  const float* W3  = (const float*)d_in[8];
  const float* b3  = (const float*)d_in[9];
  float* out = (float*)d_out;

  const int N = in_sizes[1];          // 131072
  const int groups = (N + 15) / 16;   // head groups per task

  // workspace layout (bytes)
  char* ws = (char*)d_ws;
  size_t off = 0;
  unsigned int* cnt = (unsigned int*)(ws + off);            off += 256;
  unsigned int* bucket = (unsigned int*)(ws + off);         off += (size_t)NUM_TASKS * N * 4;
  unsigned short* feats = (unsigned short*)(ws + off);      off += (size_t)N * HID * 2;
  unsigned short* wbPack = (unsigned short*)(ws + off);     off += 10 * 4 * 512 * 2;
  unsigned short* w1Pack = (unsigned short*)(ws + off);     off += (size_t)NUM_TASKS * 2 * 8 * 512 * 2;
  unsigned short* w2Pack = (unsigned short*)(ws + off);     off += (size_t)NUM_TASKS * 4 * 4 * 512 * 2;
  unsigned short* w3Pack = (unsigned short*)(ws + off);     off += (size_t)NUM_TASKS * 2 * 1 * 512 * 2;

  // stage 0: counters + weight packing
  k_zero_cnt<<<1, 32, 0, stream>>>(cnt);
  {
    long tot = 10L * 4 * 512;
    k_pack_b<<<(int)((tot + 255) / 256), 256, 0, stream>>>(
        Wb, wbPack, IN_DIM, HID, 10, 4, 1, 0, 0);
  }
  {
    long tot = (long)NUM_TASKS * 2 * 8 * 512;
    k_pack_b<<<(int)((tot + 255) / 256), 256, 0, stream>>>(
        W1, w1Pack, HID, H1D, 2, 8, NUM_TASKS, (long)HID * H1D, 2L * 8 * 512);
  }
  {
    long tot = (long)NUM_TASKS * 4 * 4 * 512;
    k_pack_b<<<(int)((tot + 255) / 256), 256, 0, stream>>>(
        W2, w2Pack, H1D, H2D, 4, 4, NUM_TASKS, (long)H1D * H2D, 4L * 4 * 512);
  }
  {
    long tot = (long)NUM_TASKS * 2 * 1 * 512;
    k_pack_b<<<(int)((tot + 255) / 256), 256, 0, stream>>>(
        W3, w3Pack, H2D, 1, 2, 1, NUM_TASKS, (long)H2D, 2L * 1 * 512);
  }

  // stage 1: bucket by task
  k_bin<<<(N + 255) / 256, 256, 0, stream>>>(ids, cnt, bucket, out, N);

  // stage 2: backbone (1 wave / 32 rows, 2 M-tiles share each B tile)
  k_backbone<<<(N + 31) / 32, 32, 0, stream>>>(x, bb, wbPack, feats, N);

  // stage 3: heads (1 wave / 16-row task-uniform group; most blocks exit early)
  k_heads<<<NUM_TASKS * groups, 32, 0, stream>>>(
      cnt, bucket, feats, w1Pack, w2Pack, w3Pack, b1, b2, b3, out, N, groups);
}
